// Model_83846351553199
// MI455X (gfx1250) — matrix operand order repaired and checked by reading, not compiled, not run on hardware
//
#include <hip/hip_runtime.h>
#include <hip/hip_bf16.h>
#include <math.h>

typedef _Float16 half_t;
typedef __attribute__((ext_vector_type(16))) _Float16 v16h;
typedef __attribute__((ext_vector_type(8)))  _Float16 v8h;
typedef __attribute__((ext_vector_type(8)))  float    v8f;

#define BB   128
#define TT   256
#define BT   (BB*TT)       // 32768
#define VV   50000
#define EE   300
#define HH   512
#define GG   2048          // 4*H
#define LL   96
#define KX   320           // E+1=301 padded to multiple of 32
#define KSPLIT 4           // split-K slices for the recurrent step GEMM

// ---------------------------------------------------------------------------
// Weight packing: dst[n*Kdst + koff + k] = (k < src_rows) ? src[k*N + n] : 0
// (transpose to [N][K] row-major so WMMA B-fragments are contiguous per lane)
// ---------------------------------------------------------------------------
__global__ void pack_bt_kernel(const float* __restrict__ src, int src_rows, int N,
                               half_t* __restrict__ dst, int Kdst, int koff, int kcount) {
    long long idx = (long long)blockIdx.x * blockDim.x + threadIdx.x;
    long long total = (long long)N * kcount;
    if (idx >= total) return;
    int k = (int)(idx % kcount);
    long long n = idx / kcount;
    float v = (k < src_rows) ? src[(long long)k * N + n] : 0.0f;
    dst[n * (long long)Kdst + koff + k] = (half_t)v;
}

// ---------------------------------------------------------------------------
// Build x_h[b*T+t][0..319] = [ emb[word_id] (300) | predicate flag | zeros ]
// ---------------------------------------------------------------------------
__global__ void embed_kernel(const int* __restrict__ word_ids,
                             const int* __restrict__ pred_ids,
                             const float* __restrict__ emb,
                             half_t* __restrict__ x_h) {
    long long idx = (long long)blockIdx.x * blockDim.x + threadIdx.x;
    if (idx >= (long long)BT * KX) return;
    int k = (int)(idx % KX);
    long long bt = idx / KX;
    float v;
    if (k < EE)        v = emb[(long long)word_ids[bt] * EE + k];
    else if (k == EE)  v = (float)pred_ids[bt];
    else               v = 0.0f;
    x_h[idx] = (half_t)v;
}

__global__ void zero_u32_kernel(unsigned* __restrict__ p, long long n) {
    long long i = (long long)blockIdx.x * blockDim.x + threadIdx.x;
    if (i < n) p[i] = 0u;
}

// ---------------------------------------------------------------------------
// WMMA fragment loads (ISA 7.12.2 layouts, wave32)
// A 16x32 f16: lane -> row = lane&15; halves 0..7 = K[kb..kb+7], 8..15 = K[kb+16..kb+23], kb = (lane>>4)*8
// B 32x16 f16 (stored transposed [N][K]): lane -> col = lane&15; halves 0..7 = K[kb..kb+7], 8..15 = K[kb+16..kb+23], kb = (lane>>4)*8
// ---------------------------------------------------------------------------
__device__ __forceinline__ v16h load_frag_a(const half_t* __restrict__ A, long long lda, int lane) {
    int row = lane & 15;
    int kb  = (lane >> 4) << 3;
    const half_t* p = A + (long long)row * lda + kb;
    v8h lo = *(const v8h*)(p);
    v8h hi = *(const v8h*)(p + 16);
    v16h r;
#pragma unroll
    for (int i = 0; i < 8; ++i) { r[i] = lo[i]; r[i + 8] = hi[i]; }
    return r;
}

__device__ __forceinline__ v16h load_frag_b(const half_t* __restrict__ Bt, long long ldb, int lane) {
    int col = lane & 15;
    int kb  = (lane >> 4) << 3;
    const half_t* p = Bt + (long long)col * ldb + kb;
    v8h lo = *(const v8h*)(p);
    v8h hi = *(const v8h*)(p + 16);
    v16h r;
#pragma unroll
    for (int i = 0; i < 8; ++i) { r[i] = lo[i]; r[i + 8] = hi[i]; }
    return r;
}

// ---------------------------------------------------------------------------
// Split-K two-phase WMMA GEMM.
//   Slice = blockIdx.y (scalar!); slice s covers K-iterations [s*per, (s+1)*per)
//   of the combined K = K1 (from A1) ++ K2 (from A2), and writes its partial
//   C tile (+ bias if given) to C + s*M*ldc. gridDim.y==1 -> plain GEMM.
//   The slice's K range is split into a phase-1 loop (A1) and a phase-2 loop
//   (A2) with *scalar* bounds, so the hot loops stay divergence-free.
// Each wave computes a 32x32 tile with 4 f32 accumulators.
// Requires M,N,K1,K2 multiples of 32 (K2 may be 0).
// ---------------------------------------------------------------------------
__global__ void gemm_wmma_kernel(const half_t* __restrict__ A1, long long lda1, int K1,
                                 const half_t* __restrict__ A2, long long lda2, int K2,
                                 const half_t* __restrict__ Bt, long long ldb,
                                 const float* __restrict__ bias,
                                 float* __restrict__ C, long long ldc,
                                 int M, int N) {
    const int lane = threadIdx.x & 31;
    const int wave = threadIdx.x >> 5;
    const int tilesN  = N >> 5;
    const int tilesMN = (M >> 5) * tilesN;
    const int tile = blockIdx.x * (blockDim.x >> 5) + wave;
    if (tile >= tilesMN) return;
    const int tm = tile / tilesN;
    const int tn = tile - tm * tilesN;
    const long long m0 = (long long)tm << 5;
    const int n0 = tn << 5;

    // Scalar (SGPR) K-range for this slice.
    const int slice  = blockIdx.y;
    const int nIter  = (K1 + K2) >> 5;
    const int per    = (nIter + gridDim.y - 1) / gridDim.y;
    int kb = slice * per * 32;
    int ke = (slice + 1) * per * 32;
    if (ke > K1 + K2) ke = K1 + K2;
    const int p1end = (ke < K1) ? ke : K1;            // phase-1: [kb, p1end)
    const int p2beg = (kb > K1) ? kb : K1;            // phase-2: [p2beg, ke)

    v8f acc00 = {}, acc01 = {}, acc10 = {}, acc11 = {};

    const half_t* b0base = Bt + (long long)n0 * ldb;
    const half_t* b1base = Bt + (long long)(n0 + 16) * ldb;

    // ---- phase 1: A1 rows, K in [kb, p1end) --------------------------------
    {
        const half_t* a0base = A1 + m0 * lda1;
        const half_t* a1base = A1 + (m0 + 16) * lda1;
#pragma unroll 4
        for (int k = kb; k < p1end; k += 32) {
            v16h a0 = load_frag_a(a0base + k, lda1, lane);
            v16h a1 = load_frag_a(a1base + k, lda1, lane);
            v16h b0 = load_frag_b(b0base + k, ldb, lane);
            v16h b1 = load_frag_b(b1base + k, ldb, lane);
            acc00 = __builtin_amdgcn_wmma_f32_16x16x32_f16(false, a0, false, b0, (short)0, acc00, false, false);
            acc01 = __builtin_amdgcn_wmma_f32_16x16x32_f16(false, a0, false, b1, (short)0, acc01, false, false);
            acc10 = __builtin_amdgcn_wmma_f32_16x16x32_f16(false, a1, false, b0, (short)0, acc10, false, false);
            acc11 = __builtin_amdgcn_wmma_f32_16x16x32_f16(false, a1, false, b1, (short)0, acc11, false, false);
        }
    }
    // ---- phase 2: A2 rows, K in [p2beg, ke) --------------------------------
    if (K2 > 0) {
        const half_t* a0base = A2 + m0 * lda2;
        const half_t* a1base = A2 + (m0 + 16) * lda2;
#pragma unroll 4
        for (int k = p2beg; k < ke; k += 32) {
            const int k2 = k - K1;
            v16h a0 = load_frag_a(a0base + k2, lda2, lane);
            v16h a1 = load_frag_a(a1base + k2, lda2, lane);
            v16h b0 = load_frag_b(b0base + k, ldb, lane);
            v16h b1 = load_frag_b(b1base + k, ldb, lane);
            acc00 = __builtin_amdgcn_wmma_f32_16x16x32_f16(false, a0, false, b0, (short)0, acc00, false, false);
            acc01 = __builtin_amdgcn_wmma_f32_16x16x32_f16(false, a0, false, b1, (short)0, acc01, false, false);
            acc10 = __builtin_amdgcn_wmma_f32_16x16x32_f16(false, a1, false, b0, (short)0, acc10, false, false);
            acc11 = __builtin_amdgcn_wmma_f32_16x16x32_f16(false, a1, false, b1, (short)0, acc11, false, false);
        }
    }

    // C/D layout: VGPR r, lane l -> row = r + (l>>4)*8, col = l&15
    float* Cs = C + (long long)slice * M * ldc;
    const int lr = (lane >> 4) << 3;
    const int lc = lane & 15;
    const float bi0 = bias ? bias[n0 + lc]      : 0.0f;
    const float bi1 = bias ? bias[n0 + 16 + lc] : 0.0f;
#pragma unroll
    for (int r = 0; r < 8; ++r) {
        long long row0 = m0 + lr + r;
        long long row1 = m0 + 16 + lr + r;
        Cs[row0 * ldc + n0 + lc]      = acc00[r] + bi0;
        Cs[row0 * ldc + n0 + 16 + lc] = acc01[r] + bi1;
        Cs[row1 * ldc + n0 + lc]      = acc10[r] + bi0;
        Cs[row1 * ldc + n0 + 16 + lc] = acc11[r] + bi1;
    }
}

// ---------------------------------------------------------------------------
// LSTM pointwise gates + mask carry (Keras order i,f,c,o); h kept as f16.
// Sums KSPLIT partial-K slabs of g (fixed order -> deterministic) + bias.
// ---------------------------------------------------------------------------
__device__ __forceinline__ float sigf(float x) { return 1.0f / (1.0f + expf(-x)); }

__global__ void lstm_gate_kernel(const float* __restrict__ g,   // [KSPLIT][BB][GG]
                                 const float* __restrict__ bias,
                                 float* __restrict__ c_st,
                                 half_t* __restrict__ h_st,
                                 half_t* __restrict__ hseq,
                                 const int* __restrict__ word_ids,
                                 int t) {
    int idx = blockIdx.x * blockDim.x + threadIdx.x;   // BB*HH
    if (idx >= BB * HH) return;
    int b = idx >> 9;
    int j = idx & (HH - 1);
    bool m = word_ids[b * TT + t] != 0;

    float pi = bias[j], pf = bias[j + HH], pc = bias[j + 2 * HH], po = bias[j + 3 * HH];
#pragma unroll
    for (int s = 0; s < KSPLIT; ++s) {
        const float* gr = g + ((long long)s * BB + b) * GG;
        pi += gr[j];
        pf += gr[j + HH];
        pc += gr[j + 2 * HH];
        po += gr[j + 3 * HH];
    }
    float gi = sigf(pi);
    float gf = sigf(pf);
    float gc = tanhf(pc);
    float go = sigf(po);

    float c_old = c_st[idx];
    half_t h_old = h_st[idx];
    float c_new = gf * c_old + gi * gc;
    float h_new = go * tanhf(c_new);
    if (!m) c_new = c_old;
    half_t h_out = m ? (half_t)h_new : h_old;
    c_st[idx] = c_new;
    h_st[idx] = h_out;
    hseq[((long long)b * TT + t) * HH + j] = h_out;
}

// ---------------------------------------------------------------------------
// In-place row softmax over 96 logits; one wave32 per row, 3 elems/lane
// ---------------------------------------------------------------------------
__global__ void softmax96_kernel(float* __restrict__ out, long long rows) {
    int lane = threadIdx.x & 31;
    int wave = threadIdx.x >> 5;
    long long row = (long long)blockIdx.x * (blockDim.x >> 5) + wave;
    if (row >= rows) return;
    float* p = out + row * LL;
    float v0 = p[lane], v1 = p[lane + 32], v2 = p[lane + 64];
    float mx = fmaxf(v0, fmaxf(v1, v2));
#pragma unroll
    for (int off = 16; off > 0; off >>= 1) mx = fmaxf(mx, __shfl_xor(mx, off, 32));
    float e0 = expf(v0 - mx), e1 = expf(v1 - mx), e2 = expf(v2 - mx);
    float s = e0 + e1 + e2;
#pragma unroll
    for (int off = 16; off > 0; off >>= 1) s += __shfl_xor(s, off, 32);
    float inv = 1.0f / s;
    p[lane] = e0 * inv; p[lane + 32] = e1 * inv; p[lane + 64] = e2 * inv;
}

// ---------------------------------------------------------------------------
static inline size_t align256(size_t x) { return (x + 255) & ~(size_t)255; }

extern "C" void kernel_launch(void* const* d_in, const int* in_sizes, int n_in,
                              void* d_out, int out_size, void* d_ws, size_t ws_size,
                              hipStream_t stream) {
    (void)in_sizes; (void)n_in; (void)out_size; (void)ws_size;
    const int*   word_ids = (const int*)d_in[0];
    const int*   pred_ids = (const int*)d_in[1];
    const float* emb      = (const float*)d_in[2];
    const float* W1 = (const float*)d_in[3];
    const float* U1 = (const float*)d_in[4];
    const float* b1 = (const float*)d_in[5];
    const float* W2 = (const float*)d_in[6];
    const float* U2 = (const float*)d_in[7];
    const float* b2 = (const float*)d_in[8];
    const float* Wd = (const float*)d_in[9];
    const float* bd = (const float*)d_in[10];
    float* out = (float*)d_out;

    // ---- workspace carve-up -------------------------------------------------
    char* ws = (char*)d_ws;
    size_t off = 0;
    half_t* Wu1t  = (half_t*)(ws + off); off = align256(off + (size_t)GG * (KX + HH) * 2); // [2048][832]
    half_t* Wu2t  = (half_t*)(ws + off); off = align256(off + (size_t)GG * (2 * HH) * 2);  // [2048][1024]
    half_t* Wdt   = (half_t*)(ws + off); off = align256(off + (size_t)LL * HH * 2);        // [96][512]
    half_t* x_h   = (half_t*)(ws + off); off = align256(off + (size_t)BT * KX * 2);        // [32768][320]
    half_t* h1seq = (half_t*)(ws + off); off = align256(off + (size_t)BT * HH * 2);
    half_t* h2seq = (half_t*)(ws + off); off = align256(off + (size_t)BT * HH * 2);
    float*  gbuf  = (float*)(ws + off);  off = align256(off + (size_t)KSPLIT * BB * GG * 4);
    half_t* h_st  = (half_t*)(ws + off); off = align256(off + (size_t)BB * HH * 2);
    float*  c_st  = (float*)(ws + off);  off = align256(off + (size_t)BB * HH * 4);

    const int TPB = 256;
    // ---- pack weights (transpose to [N][K], f16, zero-pad E+1 -> 320) ------
    {
        long long n;
        n = (long long)GG * KX;      pack_bt_kernel<<<(n + TPB - 1) / TPB, TPB, 0, stream>>>(W1, EE + 1, GG, Wu1t, KX + HH, 0,  KX);
        n = (long long)GG * HH;      pack_bt_kernel<<<(n + TPB - 1) / TPB, TPB, 0, stream>>>(U1, HH,     GG, Wu1t, KX + HH, KX, HH);
        n = (long long)GG * HH;      pack_bt_kernel<<<(n + TPB - 1) / TPB, TPB, 0, stream>>>(W2, HH,     GG, Wu2t, 2 * HH,  0,  HH);
        n = (long long)GG * HH;      pack_bt_kernel<<<(n + TPB - 1) / TPB, TPB, 0, stream>>>(U2, HH,     GG, Wu2t, 2 * HH,  HH, HH);
        n = (long long)LL * HH;      pack_bt_kernel<<<(n + TPB - 1) / TPB, TPB, 0, stream>>>(Wd, HH,     LL, Wdt,  HH,      0,  HH);
    }
    // ---- embedding gather + flag + pad -------------------------------------
    {
        long long n = (long long)BT * KX;
        embed_kernel<<<(unsigned)((n + TPB - 1) / TPB), TPB, 0, stream>>>(word_ids, pred_ids, emb, x_h);
    }

    const int stepTilesMN = (BB / 32) * (GG / 32);          // 256 tiles per slice
    const dim3 stepGrid((stepTilesMN + 3) / 4, KSPLIT);     // 4 waves/block, y = K-slice
    const int gateBlocks = (BB * HH + TPB - 1) / TPB;

    // ---- layer 1 scan -------------------------------------------------------
    zero_u32_kernel<<<(BB * HH / 2 + TPB - 1) / TPB, TPB, 0, stream>>>((unsigned*)h_st, BB * HH / 2);
    zero_u32_kernel<<<(BB * HH     + TPB - 1) / TPB, TPB, 0, stream>>>((unsigned*)c_st, BB * HH);
    for (int t = 0; t < TT; ++t) {
        gemm_wmma_kernel<<<stepGrid, 128, 0, stream>>>(
            x_h + (long long)t * KX, (long long)TT * KX, KX,
            h_st, HH, HH,
            Wu1t, KX + HH, nullptr, gbuf, GG, BB, GG);
        lstm_gate_kernel<<<gateBlocks, TPB, 0, stream>>>(gbuf, b1, c_st, h_st, h1seq, word_ids, t);
    }

    // ---- layer 2 scan -------------------------------------------------------
    zero_u32_kernel<<<(BB * HH / 2 + TPB - 1) / TPB, TPB, 0, stream>>>((unsigned*)h_st, BB * HH / 2);
    zero_u32_kernel<<<(BB * HH     + TPB - 1) / TPB, TPB, 0, stream>>>((unsigned*)c_st, BB * HH);
    for (int t = 0; t < TT; ++t) {
        gemm_wmma_kernel<<<stepGrid, 128, 0, stream>>>(
            h1seq + (long long)t * HH, (long long)TT * HH, HH,
            h_st, HH, HH,
            Wu2t, 2 * HH, nullptr, gbuf, GG, BB, GG);
        lstm_gate_kernel<<<gateBlocks, TPB, 0, stream>>>(gbuf, b2, c_st, h_st, h2seq, word_ids, t);
    }

    // ---- dense logits + softmax --------------------------------------------
    {
        const int denseTiles  = (BT / 32) * (LL / 32);     // 3072 wave-tiles
        const dim3 denseGrid((denseTiles + 3) / 4, 1);
        gemm_wmma_kernel<<<denseGrid, 128, 0, stream>>>(
            h2seq, HH, HH,
            nullptr, 0, 0,
            Wdt, HH, bd, out, LL, BT, LL);
        softmax96_kernel<<<BT / 4, 128, 0, stream>>>(out, BT);
    }
}
